// RowAttentionBlock_71751723647779
// MI455X (gfx1250) — matrix operand order repaired and checked by reading, not compiled, not run on hardware
//
#include <hip/hip_runtime.h>
#include <hip/hip_bf16.h>

// ---------------------------------------------------------------------------
// RowAttentionBlock for MI455X (gfx1250, wave32, WMMA 16x16x32 f16 -> f32 acc)
// B=1, NI=NJ=768, C_NODE=512, C_PAIR=128, H=8, C_HID=32
// ---------------------------------------------------------------------------

typedef __attribute__((ext_vector_type(16))) _Float16 v16h;
typedef __attribute__((ext_vector_type(8)))  float    v8f;

#define NI 768
#define NJ 768
#define C_NODE 512
#define C_PAIR 128
#define NHEAD 8
#define C_HID 32
#define HC 256                 // H*C_HID
#define NPAIR (768L * 768L)
#define INF_BIAS 1.0e9f

// ---------------------------------------------------------------------------
// WMMA fragment helpers (wave32). Layouts per CDNA5 ISA 7.12.2:
//  A (16x32 f16): lane l<16 -> row M=l, K = {0..7,16..23}; lane l+16 -> same
//                 row, K = {8..15,24..31}; packed 2 halves per VGPR.
//  B (32x16 f16): lane l -> col N=l&15, K as for A (same per-lane K order).
//  C/D (16x16 f32): lane l -> col l&15; rows rb..rb+7, rb = (l>=16)?8:0.
// ---------------------------------------------------------------------------
__device__ __forceinline__ v16h load_a_frag(const _Float16* A, int lda) {
  const int lane = threadIdx.x & 31;
  const int row  = lane & 15;
  const int kb   = (lane >> 4) << 3;   // 0 or 8
  const _Float16* r = A + row * lda;
  v16h a;
#pragma unroll
  for (int d = 0; d < 8; ++d) {
    const int k0 = (d < 4) ? (kb + 2 * d) : (16 + kb + 2 * (d - 4));
    a[2 * d]     = r[k0];
    a[2 * d + 1] = r[k0 + 1];
  }
  return a;
}

// B row-major [32 x ldb], take N-tile of 16 starting at B.
// Lane reads column N=lane&15, K = {kb..kb+7, 16+kb..16+kb+7}, kb = 8*(lane>>4).
__device__ __forceinline__ v16h load_b_frag_rowmajor(const _Float16* B, int ldb) {
  const int lane = threadIdx.x & 31;
  const int kb   = (lane >> 4) << 3;
  const _Float16* r = B + (lane & 15);
  v16h b;
#pragma unroll
  for (int e = 0; e < 16; ++e) b[e] = r[((e < 8) ? (kb + e) : (16 + kb + (e - 8))) * ldb];
  return b;
}

__device__ __forceinline__ v8f wmma_f16(v16h a, v16h b, v8f c) {
  return __builtin_amdgcn_wmma_f32_16x16x32_f16(false, a, false, b, (short)0, c,
                                                false, false);
}

__device__ __forceinline__ float fast_sigmoid(float x) {
  return 1.0f / (1.0f + __expf(-x));
}

// ---------------------------------------------------------------------------
// K0: f32 -> f16 weight conversion
// ---------------------------------------------------------------------------
__global__ void k_cvt_f16(const float* __restrict__ src, _Float16* __restrict__ dst, int n) {
  int i = blockIdx.x * blockDim.x + threadIdx.x;
  if (i < n) dst[i] = (_Float16)src[i];
}

// Pack W_pb (128x8) | W_pg (128x8) into a single [128 x 16] f16 B-matrix.
__global__ void k_pack_wpbg(const float* __restrict__ Wpb, const float* __restrict__ Wpg,
                            _Float16* __restrict__ dst) {
  int i = blockIdx.x * blockDim.x + threadIdx.x;
  if (i < C_PAIR * 16) {
    int r = i >> 4, c = i & 15;
    float v = (c < 8) ? Wpb[r * 8 + c] : Wpg[r * 8 + (c - 8)];
    dst[i] = (_Float16)v;
  }
}

// ---------------------------------------------------------------------------
// K1: LayerNorm of node_i / node_j. One block (256 thr) per row of 512.
// rows [0,768) -> qx (f32 kept for residual + f16), rows [768,1536) -> kx f16.
// ---------------------------------------------------------------------------
__global__ void k_ln_nodes(const float* __restrict__ node_i, const float* __restrict__ node_j,
                           const float* __restrict__ wi, const float* __restrict__ bi,
                           const float* __restrict__ wj, const float* __restrict__ bj,
                           float* __restrict__ qx_f32, _Float16* __restrict__ qx_f16,
                           _Float16* __restrict__ kx_f16) {
  const int row  = blockIdx.x;                 // 0..1535
  const bool is_i = row < NI;
  const float* x = is_i ? (node_i + (size_t)row * C_NODE)
                        : (node_j + (size_t)(row - NI) * C_NODE);
  const int t = threadIdx.x;                   // 256 threads, 2 elems each
  float x0 = x[t], x1 = x[t + 256];
  float s = x0 + x1, ss = x0 * x0 + x1 * x1;
#pragma unroll
  for (int off = 16; off; off >>= 1) {
    s  += __shfl_down(s, off, 32);
    ss += __shfl_down(ss, off, 32);
  }
  __shared__ float sh_s[8], sh_ss[8];
  const int wid = t >> 5;
  if ((t & 31) == 0) { sh_s[wid] = s; sh_ss[wid] = ss; }
  __syncthreads();
  float tot = 0.f, tot2 = 0.f;
#pragma unroll
  for (int w = 0; w < 8; ++w) { tot += sh_s[w]; tot2 += sh_ss[w]; }
  const float mean = tot * (1.0f / C_NODE);
  const float var  = tot2 * (1.0f / C_NODE) - mean * mean;
  const float rstd = rsqrtf(var + 1e-5f);
  const float* w = is_i ? wi : wj;
  const float* b = is_i ? bi : bj;
  const float y0 = (x0 - mean) * rstd * w[t]       + b[t];
  const float y1 = (x1 - mean) * rstd * w[t + 256] + b[t + 256];
  if (is_i) {
    size_t base = (size_t)row * C_NODE;
    qx_f32[base + t] = y0;       qx_f32[base + t + 256] = y1;
    qx_f16[base + t] = (_Float16)y0; qx_f16[base + t + 256] = (_Float16)y1;
  } else {
    size_t base = (size_t)(row - NI) * C_NODE;
    kx_f16[base + t] = (_Float16)y0; kx_f16[base + t + 256] = (_Float16)y1;
  }
}

// ---------------------------------------------------------------------------
// K2: projection GEMM C[768,256] = A[768,512] @ W[512,256] (f16 in, f32 acc).
// One wave per 16x16 tile; 8 waves/block; grid exactly covers 768 tiles.
// mode 0: out_h16[h][row][c] = acc*scale  (h=n/32, c=n%32)   -> q / v layout
// mode 1: out_f32[row*256+n] = sigmoid(acc + bias[n])        -> gate
// mode 2: out_h16[n][row]    = acc        (kT layout)        -> K transposed,
//         read by the score B-fragment loader with row stride NJ.
// ---------------------------------------------------------------------------
__global__ void k_proj_gemm(const _Float16* __restrict__ A, const _Float16* __restrict__ W,
                            _Float16* __restrict__ out_h16, float* __restrict__ out_f32,
                            const float* __restrict__ bias, float scale, int mode) {
  const int wave = threadIdx.x >> 5;
  const int tile = blockIdx.x * (blockDim.x >> 5) + wave;  // 0..767
  const int mt = tile >> 4, nt = tile & 15;
  const int m0 = mt * 16, n0 = nt * 16;
  v8f c = {};
#pragma unroll 4
  for (int k0 = 0; k0 < C_NODE; k0 += 32) {
    v16h a = load_a_frag(A + (size_t)m0 * C_NODE + k0, C_NODE);
    v16h b = load_b_frag_rowmajor(W + (size_t)k0 * HC + n0, HC);
    c = wmma_f16(a, b, c);
  }
  const int lane = threadIdx.x & 31;
  const int col = lane & 15, rb = (lane >> 4) << 3;
  const int n = n0 + col;
  if (mode == 0) {
    const int h = n >> 5, ch = n & 31;
#pragma unroll
    for (int r = 0; r < 8; ++r) {
      const int row = m0 + rb + r;
      out_h16[((size_t)h * NI + row) * C_HID + ch] = (_Float16)(c[r] * scale);
    }
  } else if (mode == 2) {
#pragma unroll
    for (int r = 0; r < 8; ++r) {
      const int row = m0 + rb + r;
      out_h16[(size_t)n * NJ + row] = (_Float16)(c[r] * scale);
    }
  } else {
#pragma unroll
    for (int r = 0; r < 8; ++r) {
      const int row = m0 + rb + r;
      out_f32[(size_t)row * HC + n] = fast_sigmoid(c[r] + bias[n]);
    }
  }
}

// ---------------------------------------------------------------------------
// K3: fused pair path (the 302 MB streaming kernel).
// Per wave: 16 pairs. LayerNorm(C_PAIR=128) in f32 -> f16 LDS tile [16x128],
// 4x WMMA against packed [128x16] (W_pb | W_pg), then gated bias + mask,
// written as bias[h][pair] (h = output column 0..7, pg in columns 8..15).
// ---------------------------------------------------------------------------
__global__ void k_pair_bias(const float* __restrict__ pair, const float* __restrict__ mask,
                            const float* __restrict__ lnw, const float* __restrict__ lnb,
                            const _Float16* __restrict__ Wpbg,
                            const float* __restrict__ bpb, const float* __restrict__ bpg,
                            float* __restrict__ bias_out) {
  __shared__ _Float16 sh_p[8][16 * C_PAIR];   // 32 KB (8 waves)
  __shared__ _Float16 sh_w[C_PAIR * 16];      // 4 KB
  const int t = threadIdx.x;
  for (int i = t; i < C_PAIR * 16; i += blockDim.x) sh_w[i] = Wpbg[i];

  const int wave = t >> 5, lane = t & 31;
  const long group = (long)blockIdx.x * 8 + wave;     // 16 pairs / wave
  const long p0 = group * 16;

  // Prefetch next group's data toward L2 (gfx1250 global_prefetch_b8).
  __builtin_prefetch(pair + (p0 + 128) * C_PAIR + (lane << 4), 0, 1);

  // LayerNorm: 4 lanes per pair, 32 channels per lane; 2 sub-batches of 8 pairs.
#pragma unroll
  for (int sub = 0; sub < 2; ++sub) {
    const int pp = lane & 7;        // pair within sub-batch
    const int q  = lane >> 3;       // channel quarter 0..3
    const int pr = sub * 8 + pp;    // pair row 0..15 within group
    const float* src = pair + (p0 + pr) * C_PAIR + q * 32;
    float x[32];
    float s = 0.f, ss = 0.f;
#pragma unroll
    for (int cidx = 0; cidx < 32; ++cidx) {
      const float v = src[cidx];
      x[cidx] = v; s += v; ss += v * v;
    }
    s  += __shfl_xor(s, 8, 32);   ss += __shfl_xor(ss, 8, 32);
    s  += __shfl_xor(s, 16, 32);  ss += __shfl_xor(ss, 16, 32);
    const float mean = s * (1.0f / C_PAIR);
    const float var  = ss * (1.0f / C_PAIR) - mean * mean;
    const float rstd = rsqrtf(var + 1e-5f);
#pragma unroll
    for (int cidx = 0; cidx < 32; ++cidx) {
      const int ch = q * 32 + cidx;
      const float y = (x[cidx] - mean) * rstd * lnw[ch] + lnb[ch];
      sh_p[wave][pr * C_PAIR + ch] = (_Float16)y;
    }
  }
  __syncthreads();

  v8f c = {};
#pragma unroll
  for (int k0 = 0; k0 < C_PAIR; k0 += 32) {
    v16h a = load_a_frag(&sh_p[wave][k0], C_PAIR);
    v16h b = load_b_frag_rowmajor(&sh_w[k0 * 16], 16);
    c = wmma_f16(a, b, c);
  }
  const int col = lane & 15, rb = (lane >> 4) << 3;
#pragma unroll
  for (int r = 0; r < 8; ++r) {
    const float mine  = c[r];
    const float other = __shfl_xor(mine, 8, 32);   // pb <-> pg partner column
    if (col < 8) {
      const long pidx = p0 + rb + r;
      const float gate = fast_sigmoid(other + bpg[col]);
      const float bv = (mine + bpb[col]) * gate + INF_BIAS * (mask[pidx] - 1.0f);
      bias_out[(long)col * NPAIR + pidx] = bv;
    }
  }
}

// ---------------------------------------------------------------------------
// K4: attention. One wave per (head, 16-row i-tile); 2 waves/block (48 KB LDS).
// Scores use kT (transposed K); score B-fragments are read with row stride NJ.
// Two score passes (WMMA is cheap: K=32) for max then exp/sum; f16 probs in
// LDS; P@V with two WMMA accumulators (C_HID=32 -> 2 N-tiles); epilogue
// divides by row sum, applies sigmoid gate, emits f16 [row, h*32+c].
// ---------------------------------------------------------------------------
__global__ void k_attention(const _Float16* __restrict__ q16, const _Float16* __restrict__ kT16,
                            const _Float16* __restrict__ v16, const float* __restrict__ bias,
                            const float* __restrict__ g, _Float16* __restrict__ o16) {
  __shared__ _Float16 sh_probs[2][16 * NJ];    // 48 KB
  const int wave = threadIdx.x >> 5, lane = threadIdx.x & 31;
  const int task = blockIdx.x * 2 + wave;      // 0..383
  const int h  = task / 48;
  const int it = task % 48;
  const int i0 = it * 16;
  const int col = lane & 15, rb = (lane >> 4) << 3;

  const v16h aq = load_a_frag(q16 + ((size_t)h * NI + i0) * C_HID, C_HID);
  const float* brow = bias + (long)h * NPAIR + (long)i0 * NJ;
  // kT row block for this head: kT16[(h*32 + c)*NJ + j]
  const _Float16* kTh = kT16 + (size_t)h * C_HID * NJ;

  // Pass 1: row maxima.
  float rmax[8];
#pragma unroll
  for (int r = 0; r < 8; ++r) rmax[r] = -3.0e38f;
  for (int jt = 0; jt < 48; ++jt) {
    const int j0 = jt * 16;
    v16h bk = load_b_frag_rowmajor(kTh + j0, NJ);
    v8f c = {};
    c = wmma_f16(aq, bk, c);
#pragma unroll
    for (int r = 0; r < 8; ++r) {
      const float s = c[r] + brow[(long)(rb + r) * NJ + j0 + col];
      rmax[r] = fmaxf(rmax[r], s);
    }
  }
#pragma unroll
  for (int r = 0; r < 8; ++r)
#pragma unroll
    for (int m = 1; m <= 8; m <<= 1)
      rmax[r] = fmaxf(rmax[r], __shfl_xor(rmax[r], m, 32));

  // Pass 2: exp(s - max), stage f16 probs, accumulate row sums.
  float rsum[8];
#pragma unroll
  for (int r = 0; r < 8; ++r) rsum[r] = 0.0f;
  for (int jt = 0; jt < 48; ++jt) {
    const int j0 = jt * 16;
    v16h bk = load_b_frag_rowmajor(kTh + j0, NJ);
    v8f c = {};
    c = wmma_f16(aq, bk, c);
#pragma unroll
    for (int r = 0; r < 8; ++r) {
      const float s = c[r] + brow[(long)(rb + r) * NJ + j0 + col];
      const float e = __expf(s - rmax[r]);
      rsum[r] += e;
      sh_probs[wave][(rb + r) * NJ + j0 + col] = (_Float16)e;
    }
  }
#pragma unroll
  for (int r = 0; r < 8; ++r)
#pragma unroll
    for (int m = 1; m <= 8; m <<= 1)
      rsum[r] += __shfl_xor(rsum[r], m, 32);
  __syncthreads();

  // P @ V, K-chunks of 32.
  v8f oa0 = {}, oa1 = {};
  for (int jt = 0; jt < 24; ++jt) {
    const int kk = jt * 32;
    v16h ap = load_a_frag(&sh_probs[wave][kk], NJ);
    v16h bv0 = load_b_frag_rowmajor(v16 + ((size_t)h * NJ + kk) * C_HID, C_HID);
    v16h bv1 = load_b_frag_rowmajor(v16 + ((size_t)h * NJ + kk) * C_HID + 16, C_HID);
    oa0 = wmma_f16(ap, bv0, oa0);
    oa1 = wmma_f16(ap, bv1, oa1);
  }
#pragma unroll
  for (int nt = 0; nt < 2; ++nt) {
#pragma unroll
    for (int r = 0; r < 8; ++r) {
      const int row = i0 + rb + r;
      const int n = h * C_HID + nt * 16 + col;
      const float v = (nt ? oa1[r] : oa0[r]) / rsum[r] * g[(size_t)row * HC + n];
      o16[(size_t)row * HC + n] = (_Float16)v;
    }
  }
}

// ---------------------------------------------------------------------------
// K5: output projection + residual + node mask.
// out[768,512] = qx + (o16[768,256] @ Wo16[256,512] + bo) * node_mask[row]
// ---------------------------------------------------------------------------
__global__ void k_out_gemm(const _Float16* __restrict__ o16, const _Float16* __restrict__ Wo16,
                           const float* __restrict__ bo, const float* __restrict__ qx_f32,
                           const float* __restrict__ node_mask, float* __restrict__ out) {
  const int wave = threadIdx.x >> 5;
  const int tile = blockIdx.x * (blockDim.x >> 5) + wave;  // 0..1535
  const int mt = tile >> 5, nt = tile & 31;
  const int m0 = mt * 16, n0 = nt * 16;
  v8f c = {};
#pragma unroll 4
  for (int k0 = 0; k0 < HC; k0 += 32) {
    v16h a = load_a_frag(o16 + (size_t)m0 * HC + k0, HC);
    v16h b = load_b_frag_rowmajor(Wo16 + (size_t)k0 * C_NODE + n0, C_NODE);
    c = wmma_f16(a, b, c);
  }
  const int lane = threadIdx.x & 31;
  const int col = lane & 15, rb = (lane >> 4) << 3;
  const int n = n0 + col;
#pragma unroll
  for (int r = 0; r < 8; ++r) {
    const int row = m0 + rb + r;
    out[(size_t)row * C_NODE + n] =
        qx_f32[(size_t)row * C_NODE + n] + (c[r] + bo[n]) * node_mask[row];
  }
}

// ---------------------------------------------------------------------------
// Host launcher
// ---------------------------------------------------------------------------
extern "C" void kernel_launch(void* const* d_in, const int* in_sizes, int n_in,
                              void* d_out, int out_size, void* d_ws, size_t ws_size,
                              hipStream_t stream) {
  (void)in_sizes; (void)n_in; (void)out_size; (void)ws_size;
  const float* node_i = (const float*)d_in[0];
  const float* node_j = (const float*)d_in[1];
  const float* pair   = (const float*)d_in[2];
  const float* pmask  = (const float*)d_in[3];
  const float* nmask  = (const float*)d_in[4];
  const float* ln_i_w = (const float*)d_in[5];
  const float* ln_i_b = (const float*)d_in[6];
  const float* ln_j_w = (const float*)d_in[7];
  const float* ln_j_b = (const float*)d_in[8];
  const float* ln_p_w = (const float*)d_in[9];
  const float* ln_p_b = (const float*)d_in[10];
  const float* W_pb   = (const float*)d_in[11];
  const float* b_pb   = (const float*)d_in[12];
  const float* W_pg   = (const float*)d_in[13];
  const float* b_pg   = (const float*)d_in[14];
  const float* Wq     = (const float*)d_in[15];
  const float* Wk     = (const float*)d_in[16];
  const float* Wv     = (const float*)d_in[17];
  const float* Wg     = (const float*)d_in[18];
  const float* bg     = (const float*)d_in[19];
  const float* Wo     = (const float*)d_in[20];
  const float* bo     = (const float*)d_in[21];
  float* out = (float*)d_out;

  // Workspace carve-up (all sizes 256B-aligned).
  char* ws = (char*)d_ws;
  size_t off = 0;
  float*     qx_f32 = (float*)(ws + off);     off += (size_t)NI * C_NODE * 4;
  _Float16*  qx_f16 = (_Float16*)(ws + off);  off += (size_t)NI * C_NODE * 2;
  _Float16*  kx_f16 = (_Float16*)(ws + off);  off += (size_t)NJ * C_NODE * 2;
  _Float16*  Wq16   = (_Float16*)(ws + off);  off += (size_t)C_NODE * HC * 2;
  _Float16*  Wk16   = (_Float16*)(ws + off);  off += (size_t)C_NODE * HC * 2;
  _Float16*  Wv16   = (_Float16*)(ws + off);  off += (size_t)C_NODE * HC * 2;
  _Float16*  Wg16   = (_Float16*)(ws + off);  off += (size_t)C_NODE * HC * 2;
  _Float16*  Wo16   = (_Float16*)(ws + off);  off += (size_t)HC * C_NODE * 2;
  _Float16*  Wpbg16 = (_Float16*)(ws + off);  off += (size_t)C_PAIR * 16 * 2;
  _Float16*  q16    = (_Float16*)(ws + off);  off += (size_t)NHEAD * NI * C_HID * 2;
  _Float16*  kT16   = (_Float16*)(ws + off);  off += (size_t)NHEAD * C_HID * NJ * 2;
  _Float16*  v16    = (_Float16*)(ws + off);  off += (size_t)NHEAD * NJ * C_HID * 2;
  float*     g_f32  = (float*)(ws + off);     off += (size_t)NI * HC * 4;
  float*     bias   = (float*)(ws + off);     off += (size_t)NHEAD * NPAIR * 4;
  _Float16*  o16    = (_Float16*)(ws + off);  off += (size_t)NI * HC * 2;

  // K0: weight conversion / packing.
  {
    const int n = C_NODE * HC;                      // 131072
    k_cvt_f16<<<n / 256, 256, 0, stream>>>(Wq, Wq16, n);
    k_cvt_f16<<<n / 256, 256, 0, stream>>>(Wk, Wk16, n);
    k_cvt_f16<<<n / 256, 256, 0, stream>>>(Wv, Wv16, n);
    k_cvt_f16<<<n / 256, 256, 0, stream>>>(Wg, Wg16, n);
    k_cvt_f16<<<n / 256, 256, 0, stream>>>(Wo, Wo16, n);
    k_pack_wpbg<<<(C_PAIR * 16 + 255) / 256, 256, 0, stream>>>(W_pb, W_pg, Wpbg16);
  }

  // K1: node LayerNorms.
  k_ln_nodes<<<NI + NJ, 256, 0, stream>>>(node_i, node_j, ln_i_w, ln_i_b,
                                          ln_j_w, ln_j_b, qx_f32, qx_f16, kx_f16);

  // K2: projections (48*16 = 768 tiles, 8 waves/block -> 96 blocks).
  const float qscale = 0.17677669529663687f;        // C_HID^-0.5
  k_proj_gemm<<<96, 256, 0, stream>>>(qx_f16, Wq16, q16,  nullptr, nullptr, qscale, 0);
  k_proj_gemm<<<96, 256, 0, stream>>>(kx_f16, Wk16, kT16, nullptr, nullptr, 1.0f,   2);
  k_proj_gemm<<<96, 256, 0, stream>>>(kx_f16, Wv16, v16,  nullptr, nullptr, 1.0f,   0);
  k_proj_gemm<<<96, 256, 0, stream>>>(qx_f16, Wg16, nullptr, g_f32, bg,     1.0f,   1);

  // K3: fused pair LN + gated bias + mask (36864 groups / 8 waves -> 4608 blocks).
  k_pair_bias<<<4608, 256, 0, stream>>>(pair, pmask, ln_p_w, ln_p_b,
                                        Wpbg16, b_pb, b_pg, bias);

  // K4: attention (8 heads * 48 i-tiles = 384 tasks, 2 waves/block -> 192 blocks).
  k_attention<<<192, 64, 0, stream>>>(q16, kT16, v16, bias, g_f32, o16);

  // K5: output projection + residual (48*32 = 1536 tiles, 8 waves -> 192 blocks).
  k_out_gemm<<<192, 256, 0, stream>>>(o16, Wo16, bo, qx_f32, nmask, out);
}
